// PerTokenFFN_50766513439287
// MI455X (gfx1250) — matrix operand order repaired and checked by reading, not compiled, not run on hardware
//
#include <hip/hip_runtime.h>

// ---- CDNA5 WMMA types ----
typedef __attribute__((ext_vector_type(16))) _Float16 v16h;
typedef __attribute__((ext_vector_type(8)))  _Float16 v8h;
typedef __attribute__((ext_vector_type(8)))  float    v8f;

#define B_  64
#define T_  128
#define D_  512
#define H_  2048
#define KB  256                 // H columns processed per fused block
#define XP  (D_ + 8)            // LDS pitch for X (halves): bank-conflict pad
#define HP  (KB + 8)            // LDS pitch for H block (halves)
#define NTHREADS 512            // 16 waves (wave32)

__device__ __forceinline__ v8f wmma_f16(v16h a, v16h b, v8f c) {
  // D = A(16x32 f16) * B(32x16 f16) + C(16x16 f32)
  return __builtin_amdgcn_wmma_f32_16x16x32_f16(
      /*neg_a=*/false, a, /*neg_b=*/false, b,
      /*c_mod=*/(short)0, c, /*reuse_a=*/false, /*reuse_b=*/false);
}

// A-fragment (16x32 f16, row-major source in LDS).
// Lanes 0-15: M=lane, halves 0..7 = K k0..k0+7, halves 8..15 = K k0+16..k0+23
// Lanes 16-31: M=lane-16, chunks shifted by +8 / +24. Two ds_load_b128 per lane.
__device__ __forceinline__ v16h load_a_lds(const _Float16* base, int pitch,
                                           int m0, int k0, int lane) {
  const int m = m0 + (lane & 15);
  const _Float16* p = base + m * pitch + k0 + ((lane >> 4) << 3);
  union { v16h v; v8h h[2]; } r;
  r.h[0] = *(const v8h*)(p);
  r.h[1] = *(const v8h*)(p + 16);
  return r.v;
}

// B-fragment (32x16 f16) gathered from row-major fp32 weights [ldb = row stride].
// Lane L: N = n0 + L%16; halves 0..7 = rows k0 + (L>=16?8:0) + 0..7, halves 8..15 = those + 16.
__device__ __forceinline__ v16h load_b_glb(const float* __restrict__ base, int ldb,
                                           int k0, int n0, int lane) {
  const int n = n0 + (lane & 15);
  const int k = k0 + ((lane >> 4) << 3);
  const float* p = base + (size_t)k * (size_t)ldb + n;
  v16h r;
#pragma unroll
  for (int v = 0; v < 8; ++v) {
    r[2 * v]     = (_Float16)p[0];
    r[2 * v + 1] = (_Float16)p[ldb];
    p += (v == 3 ? 10 : 2) * (size_t)ldb;
  }
  return r;
}

__global__ __launch_bounds__(NTHREADS, 1)
void PerTokenFFN_fused_kernel(const float* __restrict__ x,
                              const float* __restrict__ W1,
                              const float* __restrict__ b1,
                              const float* __restrict__ W2,
                              const float* __restrict__ b2,
                              float* __restrict__ out) {
  __shared__ __align__(16) _Float16 Xs[B_ * XP];   // X_t in f16:  ~66.5 KB
  __shared__ __align__(16) _Float16 Hs[B_ * HP];   // H col-block: ~33.8 KB

  const int t    = blockIdx.x;      // one workgroup per token
  const int tid  = threadIdx.x;
  const int wave = tid >> 5;
  const int lane = tid & 31;
  const int nlo  = lane & 15;       // column within a 16-wide strip
  const int mhi  = (lane >> 4) << 3; // 0 or 8: C/D row offset per half-wave

  const float* W1t = W1 + (size_t)t * D_ * H_;
  const float* W2t = W2 + (size_t)t * H_ * D_;
  const float* b1t = b1 + (size_t)t * H_;
  const float* b2t = b2 + (size_t)t * D_;

  // ---- stage X_t (64x512 fp32, row stride T_*D_) into LDS as f16, coalesced ----
  for (int i = tid; i < B_ * D_; i += NTHREADS) {
    const int bb = i >> 9;          // / D_
    const int dd = i & (D_ - 1);
    Xs[bb * XP + dd] = (_Float16)x[((size_t)bb * T_ + t) * D_ + dd];
  }
  __syncthreads();

  // persistent GEMM2 accumulators: wave owns out columns [wave*32, wave*32+32)
  v8f acc2[2][4];
#pragma unroll
  for (int s = 0; s < 2; ++s)
#pragma unroll
    for (int mt = 0; mt < 4; ++mt) acc2[s][mt] = (v8f){};

  for (int kb = 0; kb < H_ / KB; ++kb) {
    // ================= phase 1: H[:, kb*KB .. +KB) = gelu(X*W1 + b1) ==========
    const int n1 = kb * KB + wave * 16;        // this wave's H column strip
    v8f acc1[4];
#pragma unroll
    for (int mt = 0; mt < 4; ++mt) acc1[mt] = (v8f){};

#pragma unroll 4
    for (int ks = 0; ks < D_ / 32; ++ks) {     // K = 512 -> 16 steps
      const v16h bf = load_b_glb(W1t, H_, ks * 32, n1, lane);
#pragma unroll
      for (int mt = 0; mt < 4; ++mt) {
        const v16h af = load_a_lds(Xs, XP, mt * 16, ks * 32, lane);
        acc1[mt] = wmma_f16(af, bf, acc1[mt]);
      }
    }

    // bias + exact GeLU (erf), write f16 block to LDS
    const float bias1 = b1t[n1 + nlo];
    const int   cloc  = wave * 16 + nlo;
#pragma unroll
    for (int mt = 0; mt < 4; ++mt)
#pragma unroll
      for (int r = 0; r < 8; ++r) {
        const float v = acc1[mt][r] + bias1;
        const float g = 0.5f * v * (1.0f + erff(v * 0.70710678118654752f));
        Hs[(mt * 16 + r + mhi) * HP + cloc] = (_Float16)g;
      }
    __syncthreads();

    // ================= phase 2: out += H_block * W2[kb*KB .. , :] =============
#pragma unroll
    for (int s = 0; s < 2; ++s) {
      const int n2 = wave * 32 + s * 16;       // output column strip
#pragma unroll 2
      for (int ks = 0; ks < KB / 32; ++ks) {   // 8 steps per block
        const v16h bf = load_b_glb(W2t, D_, kb * KB + ks * 32, n2, lane);
#pragma unroll
        for (int mt = 0; mt < 4; ++mt) {
          const v16h af = load_a_lds(Hs, HP, mt * 16, ks * 32, lane);
          acc2[s][mt] = wmma_f16(af, bf, acc2[s][mt]);
        }
      }
    }
    __syncthreads();   // Hs reused next block
  }

  // ---- epilogue: + b2, store fp32 out[b][t][d] ----
#pragma unroll
  for (int s = 0; s < 2; ++s) {
    const int   d     = wave * 32 + s * 16 + nlo;
    const float bias2 = b2t[d];
#pragma unroll
    for (int mt = 0; mt < 4; ++mt)
#pragma unroll
      for (int r = 0; r < 8; ++r) {
        const int m = mt * 16 + r + mhi;
        out[((size_t)m * T_ + t) * D_ + d] = acc2[s][mt][r] + bias2;
      }
  }
}

extern "C" void kernel_launch(void* const* d_in, const int* in_sizes, int n_in,
                              void* d_out, int out_size, void* d_ws, size_t ws_size,
                              hipStream_t stream) {
  (void)in_sizes; (void)n_in; (void)out_size; (void)d_ws; (void)ws_size;
  const float* x  = (const float*)d_in[0];
  const float* W1 = (const float*)d_in[1];
  const float* b1 = (const float*)d_in[2];
  const float* W2 = (const float*)d_in[3];
  const float* b2 = (const float*)d_in[4];
  float* out = (float*)d_out;

  hipLaunchKernelGGL(PerTokenFFN_fused_kernel, dim3(T_), dim3(NTHREADS), 0, stream,
                     x, W1, b1, W2, b2, out);
}
